// ScanLayer_80144089743475
// MI455X (gfx1250) — compile-verified, matrix operand order checked by reading, not run on hardware
//
#include <hip/hip_runtime.h>

// ---------------- problem constants ----------------
constexpr int BS = 4, N = 2048, C = 8, D = 128, E = 256, E2 = 512;
constexpr int T = 64;          // time chunk
constexpr int NC = N / T;      // 32 chunks
constexpr int EPAD1 = E + 4;   // f32 LDS row stride for a/b staging
constexpr int DPAD  = D + 8;   // bf16 LDS row stride for x_norm
constexpr int EPAD2 = E + 8;   // bf16 LDS row stride for h staging

typedef __attribute__((ext_vector_type(16))) __bf16 v16bf;
typedef __attribute__((ext_vector_type(8)))  float  v8f;

// round-to-nearest-even f32 -> bf16 (bit pattern)
__device__ __forceinline__ unsigned short f2bf(float f) {
    unsigned u = __float_as_uint(f);
    unsigned r = u + 0x7FFFu + ((u >> 16) & 1u);
    return (unsigned short)(r >> 16);
}

// fast sigmoid / silu: v_exp_f32 + v_rcp_f32 (no IEEE divide expansion)
__device__ __forceinline__ float fast_sigmoid(float v) {
    return __builtin_amdgcn_rcpf(1.0f + __expf(-v));
}
__device__ __forceinline__ float fast_silu(float v) {
    return v * __builtin_amdgcn_rcpf(1.0f + __expf(-v));
}

// Build a 16x(32) bf16 WMMA fragment from two contiguous 8-element runs
// (k..k+7 and k+16..k+23), per the CDNA5 16-bit A/B VGPR layout.
__device__ __forceinline__ v16bf ld_frag16(const unsigned short* p) {
    uint4 lo = *(const uint4*)(p);
    uint4 hi = *(const uint4*)(p + 16);
    v16bf f;
    ((uint4*)&f)[0] = lo;
    ((uint4*)&f)[1] = hi;
    return f;
}

// ---------------- kernel 0: weights -> bf16, [n][k] transposed ----------------
__global__ void __launch_bounds__(256) k_convert(
        const float* __restrict__ w_in, const float* __restrict__ w_out,
        unsigned short* __restrict__ w_inT, unsigned short* __restrict__ w_outT) {
    const int NIN  = C * D * E2;   // 524288
    const int NOUT = C * E * D;    // 262144
    int g = blockIdx.x * blockDim.x + threadIdx.x;
    if (g < NIN) {
        int k = g % D, n = (g / D) % E2, c = g / (D * E2);       // w_inT[c][n][k]
        w_inT[g] = f2bf(w_in[((size_t)c * D + k) * E2 + n]);
    } else if (g < NIN + NOUT) {
        int gg = g - NIN;
        int e = gg % E, d = (gg / E) % D, c = gg / (E * D);      // w_outT[c][d][e]
        w_outT[gg] = f2bf(w_out[((size_t)c * E + e) * D + d]);
    }
}

// ---------------- kernel 1: LN + in-GEMM + activations + chunk-local scan ----
__global__ void __launch_bounds__(256) k_phase1(
        const float* __restrict__ x, const float* __restrict__ h_prev,
        const float* __restrict__ gamma, const float* __restrict__ beta,
        const float* __restrict__ b_in, const unsigned short* __restrict__ w_inT,
        float* __restrict__ Acum, float* __restrict__ Hloc,
        float* __restrict__ A_end, float* __restrict__ H_end) {
    extern __shared__ char smem[];
    float* aS = (float*)smem;                                   // [T][EPAD1]
    float* bS = aS + T * EPAD1;                                 // [T][EPAD1]
    unsigned short* xnS = (unsigned short*)(bS + T * EPAD1);    // [T][DPAD] bf16

    const int chunk = blockIdx.x, c = blockIdx.y, b = blockIdx.z;
    const int tid = threadIdx.x, wave = tid >> 5, lane = tid & 31;
    const int laneLo = lane & 15, hi = lane >> 4;
    const int n0 = chunk * T;

    // ---- grouped LayerNorm over D=128, one wave per 8 rows, float4 per lane
    float4 g4  = ((const float4*)(gamma + c * D))[lane];
    float4 be4 = ((const float4*)(beta  + c * D))[lane];
    for (int rr = 0; rr < 8; ++rr) {
        int t = wave * 8 + rr;
        const float* xr = x + (((size_t)b * N + n0 + t) * C + c) * D;
        float4 v = ((const float4*)xr)[lane];
        float s1 = v.x + v.y + v.z + v.w;
        float s2 = v.x * v.x + v.y * v.y + v.z * v.z + v.w * v.w;
        for (int off = 16; off; off >>= 1) {
            s1 += __shfl_xor(s1, off, 32);
            s2 += __shfl_xor(s2, off, 32);
        }
        float mean = s1 * (1.0f / D);
        float var  = s2 * (1.0f / D) - mean * mean;
        float rstd = __builtin_amdgcn_rsqf(var + 1e-5f);   // v_rsq_f32
        float y0 = (v.x - mean) * rstd * g4.x + be4.x;
        float y1 = (v.y - mean) * rstd * g4.y + be4.y;
        float y2 = (v.z - mean) * rstd * g4.z + be4.z;
        float y3 = (v.w - mean) * rstd * g4.w + be4.w;
        unsigned u0 = (unsigned)f2bf(y0) | ((unsigned)f2bf(y1) << 16);
        unsigned u1 = (unsigned)f2bf(y2) | ((unsigned)f2bf(y3) << 16);
        *(uint2*)(xnS + t * DPAD + lane * 4) = make_uint2(u0, u1);
    }
    __syncthreads();

    // ---- in-GEMM: [64 x 128] x [128 x 512], bf16 WMMA, f32 accum
    const int m0 = (wave & 3) * 16;     // row block
    const int ch = wave >> 2;           // column interleave
    v16bf afrag[4];
    for (int kk = 0; kk < 4; ++kk)
        afrag[kk] = ld_frag16(xnS + (m0 + laneLo) * DPAD + kk * 32 + hi * 8);

    for (int cb = 0; cb < 16; ++cb) {
        int nb = cb * 2 + ch;
        int ncol = nb * 16 + laneLo;
        const unsigned short* wb = w_inT + ((size_t)c * E2 + ncol) * D + hi * 8;
        v8f acc = {0.f, 0.f, 0.f, 0.f, 0.f, 0.f, 0.f, 0.f};
        for (int kk = 0; kk < 4; ++kk) {
            v16bf bfrag = ld_frag16(wb + kk * 32);
            acc = __builtin_amdgcn_wmma_f32_16x16x32_bf16(
                false, afrag[kk], false, bfrag, (short)0, acc, false, false);
        }
        float bias = b_in[c * E2 + ncol];
        // wave-uniform split: force a scalar branch (no EXEC juggling)
        if (__builtin_amdgcn_readfirstlane(nb) < 16) {   // sigmoid gate -> aS
            for (int r = 0; r < 8; ++r) {
                int t = m0 + r + 8 * hi;
                aS[t * EPAD1 + ncol] = fast_sigmoid(acc[r] + bias);
            }
        } else {                                         // silu input -> bS
            for (int r = 0; r < 8; ++r) {
                int t = m0 + r + 8 * hi;
                bS[t * EPAD1 + (ncol - E)] = fast_silu(acc[r] + bias);
            }
        }
    }
    __syncthreads();

    // ---- chunk-local sequential scan, thread = channel e (t=0 peeled)
    {
        int e = tid;
        size_t base = (((size_t)b * C + c) * N + n0) * E + e;
        float a0 = aS[e];
        float b0 = bS[e];
        if (chunk == 0) b0 = fmaf(a0, h_prev[((size_t)b * C + c) * E + e], b0);
        float A = a0, H = b0;
        Acum[base] = A;
        Hloc[base] = H;
        for (int t = 1; t < T; ++t) {
            float a  = aS[t * EPAD1 + e];
            float bb = bS[t * EPAD1 + e];
            H = fmaf(a, H, bb);
            A *= a;
            Acum[base + (size_t)t * E] = A;
            Hloc[base + (size_t)t * E] = H;
        }
        size_t se = (((size_t)b * C + c) * NC + chunk) * E + e;
        A_end[se] = A;
        H_end[se] = H;
    }
}

// ---------------- kernel 2: inter-chunk scan (32 steps / channel) ------------
__global__ void __launch_bounds__(256) k_phase2(
        const float* __restrict__ A_end, const float* __restrict__ H_end,
        float* __restrict__ carry_in, float* __restrict__ h_last) {
    int g = blockIdx.x * blockDim.x + threadIdx.x;   // 0 .. BS*C*E-1 = 8191
    int e = g & (E - 1);
    int bc = g >> 8;                                 // b*C + c
    float carry = 0.0f;
    size_t base = (size_t)bc * NC * E + e;
    for (int k = 0; k < NC; ++k) {
        size_t idx = base + (size_t)k * E;
        carry_in[idx] = carry;
        carry = fmaf(A_end[idx], carry, H_end[idx]);
    }
    h_last[g] = carry;   // [BS][C][E] flat == g
}

// ---------------- kernel 3: apply carry + out-GEMM + residual ----------------
__global__ void __launch_bounds__(256) k_phase3(
        const float* __restrict__ x, const float* __restrict__ b_out,
        const unsigned short* __restrict__ w_outT,
        const float* __restrict__ Acum, const float* __restrict__ Hloc,
        const float* __restrict__ carry_in, float* __restrict__ out) {
    extern __shared__ char smem[];
    unsigned short* hS = (unsigned short*)smem;      // [T][EPAD2] bf16
    const int chunk = blockIdx.x, c = blockIdx.y, b = blockIdx.z;
    const int tid = threadIdx.x, wave = tid >> 5, lane = tid & 31;
    const int laneLo = lane & 15, hi = lane >> 4;
    const int n0 = chunk * T;

    {   // h = Acum * carry + Hloc, stage as bf16 [t][e]
        int e = tid;
        float carry = carry_in[(((size_t)b * C + c) * NC + chunk) * E + e];
        size_t base = (((size_t)b * C + c) * N + n0) * E + e;
        for (int t = 0; t < T; ++t) {
            float h = fmaf(Acum[base + (size_t)t * E], carry, Hloc[base + (size_t)t * E]);
            hS[t * EPAD2 + e] = f2bf(h);
        }
    }
    __syncthreads();

    // out-GEMM: [64 x 256] x [256 x 128]
    const int m0 = (wave & 3) * 16;
    v16bf afrag[8];
    for (int kk = 0; kk < 8; ++kk)
        afrag[kk] = ld_frag16(hS + (m0 + laneLo) * EPAD2 + kk * 32 + hi * 8);

    for (int j = 0; j < 4; ++j) {
        int nb = (wave >> 2) * 4 + j;
        int d = nb * 16 + laneLo;
        const unsigned short* wb = w_outT + ((size_t)c * D + d) * E + hi * 8;
        v8f acc = {0.f, 0.f, 0.f, 0.f, 0.f, 0.f, 0.f, 0.f};
        for (int kk = 0; kk < 8; ++kk) {
            v16bf bfrag = ld_frag16(wb + kk * 32);
            acc = __builtin_amdgcn_wmma_f32_16x16x32_bf16(
                false, afrag[kk], false, bfrag, (short)0, acc, false, false);
        }
        float bo = b_out[c * D + d];
        for (int r = 0; r < 8; ++r) {
            int t = m0 + r + 8 * hi;
            size_t oi = (((size_t)b * N + n0 + t) * C + c) * D + d;
            out[oi] = acc[r] + bo + x[oi];
        }
    }
}

// ---------------- launcher ----------------
extern "C" void kernel_launch(void* const* d_in, const int* in_sizes, int n_in,
                              void* d_out, int out_size, void* d_ws, size_t ws_size,
                              hipStream_t stream) {
    (void)in_sizes; (void)n_in; (void)out_size; (void)ws_size;
    const float* x      = (const float*)d_in[0];
    const float* h_prev = (const float*)d_in[1];
    const float* gamma  = (const float*)d_in[2];
    const float* beta   = (const float*)d_in[3];
    const float* w_in   = (const float*)d_in[4];
    const float* b_in   = (const float*)d_in[5];
    const float* w_out  = (const float*)d_in[6];
    const float* b_out  = (const float*)d_in[7];
    float* out = (float*)d_out;
    float* h_last = out + (size_t)BS * N * C * D;

    // workspace carve (256B aligned)
    char* ws = (char*)d_ws;
    size_t off = 0;
    auto carve = [&](size_t bytes) {
        void* p = ws + off;
        off = (off + bytes + 255) & ~(size_t)255;
        return p;
    };
    unsigned short* w_inT  = (unsigned short*)carve((size_t)C * D * E2 * 2);
    unsigned short* w_outT = (unsigned short*)carve((size_t)C * E * D * 2);
    float* Acum     = (float*)carve((size_t)BS * C * N * E * 4);
    float* Hloc     = (float*)carve((size_t)BS * C * N * E * 4);
    float* A_end    = (float*)carve((size_t)BS * C * NC * E * 4);
    float* H_end    = (float*)carve((size_t)BS * C * NC * E * 4);
    float* carry_in = (float*)carve((size_t)BS * C * NC * E * 4);

    const int NCONV = C * D * E2 + C * E * D;
    k_convert<<<(NCONV + 255) / 256, 256, 0, stream>>>(w_in, w_out, w_inT, w_outT);

    size_t smem1 = (size_t)(2 * T * EPAD1) * 4 + (size_t)T * DPAD * 2;  // ~147 KB
    k_phase1<<<dim3(NC, C, BS), 256, smem1, stream>>>(
        x, h_prev, gamma, beta, b_in, w_inT, Acum, Hloc, A_end, H_end);

    k_phase2<<<(BS * C * E) / 256, 256, 0, stream>>>(A_end, H_end, carry_in, h_last);

    size_t smem3 = (size_t)T * EPAD2 * 2;                               // ~33 KB
    k_phase3<<<dim3(NC, C, BS), 256, smem3, stream>>>(
        x, b_out, w_outT, Acum, Hloc, carry_in, out);
}
